// RNNmodel_69483980914822
// MI455X (gfx1250) — compile-verified, matrix operand order checked by reading, not run on hardware
//
#include <hip/hip_runtime.h>
#include <hip/hip_bf16.h>
#include <math.h>

// Problem constants (match reference)
#define B_ 64
#define T_ 1024
#define I_ 256
#define H_ 1024
#define O_ 256
#define TH_ ((size_t)T_ * (size_t)H_)   // row stride (elements) between batches in [B,T,H]

typedef __attribute__((ext_vector_type(16))) __bf16 v16bf;
typedef __attribute__((ext_vector_type(8)))  __bf16 v8bf;
typedef __attribute__((ext_vector_type(8)))  float  v8f;

// ---------------------------------------------------------------------------
// Fragment loader: 16x32 bf16 tile from a row-major matrix.
// Per CDNA5 ISA (CDNA5 ISA, "16-bit A-Matrix 16x32"):
//   lanes 0-15  : row = lane,    K = {0..7} in V0-3, {16..23} in V4-7
//   lanes 16-31 : row = lane-16, K = {8..15},        {24..31}
// B (32x16, needed as W^T so element (k,n)=W[n][k]) mirrors this with lane = N.
// Each half is one contiguous 16B chunk -> global_load_b128.
// ---------------------------------------------------------------------------
__device__ __forceinline__ v16bf load_frag16x32(const __hip_bfloat16* frag_base,
                                                size_t row_stride, int lane) {
    const int l  = lane & 15;
    const int hi = (lane >> 4) & 1;
    const __hip_bfloat16* p = frag_base + (size_t)l * row_stride + (size_t)(hi * 8);
    v8bf c0 = *reinterpret_cast<const v8bf*>(p);        // K chunk {0..7}+hi*8
    v8bf c1 = *reinterpret_cast<const v8bf*>(p + 16);   // K chunk {16..23}+hi*8
    return __builtin_shufflevector(c0, c1, 0,1,2,3,4,5,6,7,8,9,10,11,12,13,14,15);
}

// ---------------------------------------------------------------------------
// Generic bf16 WMMA GEMM, f32 out:  C[M,N] = A[M,K] * Bw[N,K]^T + bias[N]
// Register blocking: each wave owns 1 M-tile x 4 N-tiles. The A fragment is
// loaded once per k-step and reused by 4 WMMAs (4 independent accumulator
// chains -> A-side L2 traffic /4 and better XDL pipelining).
// Grid sized so waves == Mtiles * (Ntiles/4) exactly.
// ---------------------------------------------------------------------------
__global__ void wmma_gemm_bias_f32(const __hip_bfloat16* __restrict__ A,
                                   const __hip_bfloat16* __restrict__ Bw,
                                   const float* __restrict__ bias,
                                   float* __restrict__ C,
                                   int Mtiles, int NtileGroups, int K,
                                   int ldA, int ldB, int ldC) {
    const int lane  = threadIdx.x & 31;
    const int gwave = blockIdx.x * (blockDim.x >> 5) + (threadIdx.x >> 5);
    if (gwave >= Mtiles * NtileGroups) return;          // wave-uniform; EXEC stays full
    const int mt  = gwave / NtileGroups;
    const int ntg = gwave % NtileGroups;                // group of 4 N-tiles

    const __hip_bfloat16* Abase = A  + (size_t)mt * 16 * (size_t)ldA;
    const __hip_bfloat16* Bbase = Bw + (size_t)ntg * 64 * (size_t)ldB;

    v8f acc0 = {0.f,0.f,0.f,0.f,0.f,0.f,0.f,0.f};
    v8f acc1 = acc0, acc2 = acc0, acc3 = acc0;

    #pragma unroll 2
    for (int k0 = 0; k0 < K; k0 += 32) {
        const v16bf a  = load_frag16x32(Abase + k0, (size_t)ldA, lane);
        const v16bf b0 = load_frag16x32(Bbase + k0,                        (size_t)ldB, lane);
        const v16bf b1 = load_frag16x32(Bbase + (size_t)16 * ldB + k0,     (size_t)ldB, lane);
        const v16bf b2 = load_frag16x32(Bbase + (size_t)32 * ldB + k0,     (size_t)ldB, lane);
        const v16bf b3 = load_frag16x32(Bbase + (size_t)48 * ldB + k0,     (size_t)ldB, lane);
        acc0 = __builtin_amdgcn_wmma_f32_16x16x32_bf16(false, a, false, b0, (short)0, acc0, false, false);
        acc1 = __builtin_amdgcn_wmma_f32_16x16x32_bf16(false, a, false, b1, (short)0, acc1, false, false);
        acc2 = __builtin_amdgcn_wmma_f32_16x16x32_bf16(false, a, false, b2, (short)0, acc2, false, false);
        acc3 = __builtin_amdgcn_wmma_f32_16x16x32_bf16(false, a, false, b3, (short)0, acc3, false, false);
    }

    // C/D layout: VGPR r -> M = r + 8*(lane>=16), N = lane&15
    const int hi   = (lane >> 4) & 1;
    const int ncol = (lane & 15);
    v8f accs[4] = {acc0, acc1, acc2, acc3};
    #pragma unroll
    for (int j = 0; j < 4; ++j) {
        const int col = ntg * 64 + j * 16 + ncol;
        const float bv = bias ? bias[col] : 0.f;
        #pragma unroll
        for (int r = 0; r < 8; ++r) {
            const int row = mt * 16 + r + hi * 8;
            C[(size_t)row * (size_t)ldC + col] = accs[j][r] + bv;
        }
    }
}

// ---------------------------------------------------------------------------
// One recurrence step:  h_t = tanh(xp_t + h_{t-1} * W_hh^T)
// Each wave owns ALL 4 M-tiles (the full B=64 batch) x 1 N-tile, so the
// W_hh fragment is loaded once per k-step and reused by 4 WMMAs: W_hh (2 MB,
// L2-resident) is swept exactly once per timestep. 64 waves = 8 blocks.
// hprev rows strided by ldPrev (H for h0 buffer, T*H inside hs[B,T,H]).
// ---------------------------------------------------------------------------
__global__ void rnn_step_kernel(const __hip_bfloat16* __restrict__ hprev, size_t ldPrev,
                                const float* __restrict__ xpt,
                                const __hip_bfloat16* __restrict__ Whh,   // [H,H] row-major
                                __hip_bfloat16* __restrict__ hcur,
                                float* __restrict__ hfinal) {             // [B,H] or null
    const int lane  = threadIdx.x & 31;
    const int nt    = blockIdx.x * (blockDim.x >> 5) + (threadIdx.x >> 5); // 0..63

    const __hip_bfloat16* Bbase = Whh + (size_t)nt * 16 * (size_t)H_;

    v8f acc0 = {0.f,0.f,0.f,0.f,0.f,0.f,0.f,0.f};
    v8f acc1 = acc0, acc2 = acc0, acc3 = acc0;

    #pragma unroll 2
    for (int k0 = 0; k0 < H_; k0 += 32) {
        const v16bf b  = load_frag16x32(Bbase + k0, (size_t)H_, lane);
        const v16bf a0 = load_frag16x32(hprev + k0,                     ldPrev, lane);
        const v16bf a1 = load_frag16x32(hprev + (size_t)16 * ldPrev + k0, ldPrev, lane);
        const v16bf a2 = load_frag16x32(hprev + (size_t)32 * ldPrev + k0, ldPrev, lane);
        const v16bf a3 = load_frag16x32(hprev + (size_t)48 * ldPrev + k0, ldPrev, lane);
        acc0 = __builtin_amdgcn_wmma_f32_16x16x32_bf16(false, a0, false, b, (short)0, acc0, false, false);
        acc1 = __builtin_amdgcn_wmma_f32_16x16x32_bf16(false, a1, false, b, (short)0, acc1, false, false);
        acc2 = __builtin_amdgcn_wmma_f32_16x16x32_bf16(false, a2, false, b, (short)0, acc2, false, false);
        acc3 = __builtin_amdgcn_wmma_f32_16x16x32_bf16(false, a3, false, b, (short)0, acc3, false, false);
    }

    const int hi  = (lane >> 4) & 1;
    const int col = nt * 16 + (lane & 15);
    v8f accs[4] = {acc0, acc1, acc2, acc3};
    #pragma unroll
    for (int m = 0; m < 4; ++m) {
        #pragma unroll
        for (int r = 0; r < 8; ++r) {
            const int b = m * 16 + r + hi * 8;       // batch index 0..63
            const size_t off = (size_t)b * TH_ + col;
            const float v = tanhf(accs[m][r] + xpt[off]);
            hcur[off] = __float2bfloat16(v);
            if (hfinal) hfinal[b * H_ + col] = v;    // f32 final hidden state
        }
    }
}

// ---------------------------------------------------------------------------
// Elementwise helpers
// ---------------------------------------------------------------------------
__global__ void f32_to_bf16_kernel(const float* __restrict__ src,
                                   __hip_bfloat16* __restrict__ dst, size_t n) {
    size_t i = (size_t)blockIdx.x * blockDim.x + threadIdx.x;
    const size_t stride = (size_t)gridDim.x * blockDim.x;
    for (; i < n; i += stride) dst[i] = __float2bfloat16(src[i]);
}

__global__ void bias_sum_kernel(const float* __restrict__ a,
                                const float* __restrict__ b,
                                float* __restrict__ out, int n) {
    const int i = blockIdx.x * blockDim.x + threadIdx.x;
    if (i < n) out[i] = a[i] + b[i];
}

// ---------------------------------------------------------------------------
// Host-side orchestration
// ---------------------------------------------------------------------------
extern "C" void kernel_launch(void* const* d_in, const int* in_sizes, int n_in,
                              void* d_out, int out_size, void* d_ws, size_t ws_size,
                              hipStream_t stream) {
    (void)in_sizes; (void)n_in; (void)out_size; (void)ws_size;

    const float* inputs = (const float*)d_in[0];   // [B,T,I]
    const float* h0     = (const float*)d_in[1];   // [B,H]
    /* d_in[2] = steps (dead in reference) */
    const float* W_ih   = (const float*)d_in[3];   // [H,I]
    const float* W_hh   = (const float*)d_in[4];   // [H,H]
    const float* b_ih   = (const float*)d_in[5];   // [H]
    const float* b_hh   = (const float*)d_in[6];   // [H]
    const float* W_out  = (const float*)d_in[7];   // [O,H]
    const float* b_out  = (const float*)d_in[8];   // [O]

    // Workspace carve-out (all sizes already 256B-aligned)
    char* ws = (char*)d_ws;
    size_t off = 0;
    auto take = [&](size_t bytes) -> char* {
        char* p = ws + off;
        off += (bytes + 255) & ~(size_t)255;
        return p;
    };
    __hip_bfloat16* in_bf   = (__hip_bfloat16*)take((size_t)B_*T_*I_*2); //  32 MB
    __hip_bfloat16* h0_bf   = (__hip_bfloat16*)take((size_t)B_*H_*2);
    __hip_bfloat16* Wih_bf  = (__hip_bfloat16*)take((size_t)H_*I_*2);
    __hip_bfloat16* Whh_bf  = (__hip_bfloat16*)take((size_t)H_*H_*2);
    __hip_bfloat16* Wout_bf = (__hip_bfloat16*)take((size_t)O_*H_*2);
    float*          bsum    = (float*)take((size_t)H_*4);
    float*          xp      = (float*)take((size_t)B_*T_*H_*4);          // 256 MB
    __hip_bfloat16* hs      = (__hip_bfloat16*)take((size_t)B_*T_*H_*2); // 128 MB

    float* out    = (float*)d_out;                 // [B,T,O]
    float* hidden = out + (size_t)B_*T_*O_;        // [1,B,H]

    // --- 1) precision conversions + bias fold -----------------------------
    f32_to_bf16_kernel<<<4096, 256, 0, stream>>>(inputs, in_bf, (size_t)B_*T_*I_);
    f32_to_bf16_kernel<<<256,  256, 0, stream>>>(h0,     h0_bf, (size_t)B_*H_);
    f32_to_bf16_kernel<<<1024, 256, 0, stream>>>(W_ih,   Wih_bf, (size_t)H_*I_);
    f32_to_bf16_kernel<<<4096, 256, 0, stream>>>(W_hh,   Whh_bf, (size_t)H_*H_);
    f32_to_bf16_kernel<<<1024, 256, 0, stream>>>(W_out,  Wout_bf, (size_t)O_*H_);
    bias_sum_kernel<<<4, 256, 0, stream>>>(b_ih, b_hh, bsum, H_);

    // --- 2) x_proj = inputs @ W_ih^T + (b_ih + b_hh)   [BT,H] f32 ---------
    {
        const int Mtiles = (B_*T_) / 16;   // 4096
        const int Ngroups = (H_ / 16) / 4; // 16 groups of 4 N-tiles
        const int blocks = (Mtiles * Ngroups) / 8;   // 8192 blocks x 8 waves
        wmma_gemm_bias_f32<<<blocks, 256, 0, stream>>>(in_bf, Wih_bf, bsum, xp,
                                                       Mtiles, Ngroups, I_, I_, I_, H_);
    }

    // --- 3) sequential scan: 1024 dependent GEMM+tanh launches ------------
    for (int t = 0; t < T_; ++t) {
        const __hip_bfloat16* hprev = (t == 0) ? h0_bf : (hs + (size_t)(t-1) * H_);
        const size_t ldPrev         = (t == 0) ? (size_t)H_ : TH_;
        rnn_step_kernel<<<8, 256, 0, stream>>>(hprev, ldPrev,
                                               xp + (size_t)t * H_,
                                               Whh_bf,
                                               hs + (size_t)t * H_,
                                               (t == T_ - 1) ? hidden : nullptr);
    }

    // --- 4) output = hs @ W_out^T + b_out   [BT,O] f32 --------------------
    {
        const int Mtiles = (B_*T_) / 16;   // 4096
        const int Ngroups = (O_ / 16) / 4; // 4 groups
        const int blocks = (Mtiles * Ngroups) / 8;   // 2048 blocks x 8 waves
        wmma_gemm_bias_f32<<<blocks, 256, 0, stream>>>(hs, Wout_bf, b_out, out,
                                                       Mtiles, Ngroups, H_, H_, H_, O_);
    }
}
